// GraphConv2d_5128190952113
// MI455X (gfx1250) — matrix operand order repaired and checked by reading, not compiled, not run on hardware
//
#include <hip/hip_runtime.h>

// GraphConv (EdgeConv-style) fused: gather -> 1x1 conv (WMMA f16->f32) -> BN
// (batch stats, two-pass recompute) -> ReLU -> max over K.
// Sizes fixed by the reference problem.

typedef __attribute__((ext_vector_type(16))) _Float16 v16h;
typedef __attribute__((ext_vector_type(8)))  _Float16 v8h;
typedef __attribute__((ext_vector_type(8)))  float    v8f;

#define B_    8
#define C_    64
#define N_    8192
#define K_    20
#define COUT  64
#define C2    128                 // 2*C = GEMM reduction dim
#define NT    16                  // n-rows per workgroup
#define SAMP  (NT * K_)           // 320 samples per workgroup
#define TILES (SAMP / 16)         // 20 sample-tiles of 16
#define FSTR  136                 // padded LDS feat row stride in halves (272B)

static __device__ __forceinline__ v8f wmma_f16(v16h a, v16h b, v8f c) {
  // 8 args: (neg_a, A, neg_b, B, c_mod, C, reuse_a, reuse_b)
  return __builtin_amdgcn_wmma_f32_16x16x32_f16(false, a, false, b,
                                                (short)0, c, false, false);
}

// A-matrix (16x32 f16) fragment for output-channel tile `wave`, k-chunk `kc`.
// Layout (ISA 7.12.2): lanes 0-15 hold M=0..15; halves 0..7 -> K=0..7 (lo) /
// K=8..15 (hi lanes); halves 8..15 -> K=16..23 / K=24..31.
static __device__ __forceinline__ v16h load_a_frag(const float* __restrict__ w,
                                                   int wave, int lane, int kc) {
  int m = lane & 15;
  const float* row = w + (wave * 16 + m) * C2 + kc * 32;
  int off = (lane < 16) ? 0 : 8;
  v16h a;
#pragma unroll
  for (int i = 0; i < 8; ++i) {
    a[i]     = (_Float16)row[off + i];
    a[8 + i] = (_Float16)row[16 + off + i];
  }
  return a;
}

// B-matrix (32x16 f16) fragment from LDS feat tile.
// Lanes 0-15: N=lane, K=kc*32+0..7,16..23 ; lanes 16-31: N=lane-16, K=kc*32+8..15,24..31.
static __device__ __forceinline__ v16h load_b_frag(const _Float16* __restrict__ feat,
                                                   int lane, int kc) {
  int n = lane & 15;
  int koff = kc * 32 + ((lane < 16) ? 0 : 8);
  const v8h* p = (const v8h*)(feat + n * FSTR + koff);  // 16B aligned
  v8h b0 = p[0];
  v8h b1 = p[2];
  v16h b;
#pragma unroll
  for (int i = 0; i < 8; ++i) {
    b[i]     = b0[i];
    b[8 + i] = b1[i];
  }
  return b;
}

// Gather one 16-sample feat tile into LDS as f16 [16 samples][128 channels].
// thread t: sample j = t&15, channel group cg = t>>4 (8 channels each).
static __device__ __forceinline__ void gather_tile(_Float16* __restrict__ featLDS,
                                                   const float* __restrict__ x,
                                                   const int* __restrict__ ei,
                                                   int b, int n0, int tile, int tid) {
  int j  = tid & 15;
  int cg = tid >> 4;
  int s  = tile * 16 + j;
  int nl = s / K_;
  int k  = s - nl * K_;
  int n  = n0 + nl;
  int ibase = ((b * N_) + n) * K_ + k;
  int idx_j = ei[ibase];                 // edge_index[0] -> x_j
  int idx_i = ei[B_ * N_ * K_ + ibase];  // edge_index[1] -> x_i
  const float* xb = x + (size_t)b * C_ * N_;
#pragma unroll
  for (int u = 0; u < 8; ++u) {
    int c = cg * 8 + u;
    float xi = xb[(size_t)c * N_ + idx_i];
    float xj = xb[(size_t)c * N_ + idx_j];
    featLDS[j * FSTR + c]      = (_Float16)xi;
    featLDS[j * FSTR + C_ + c] = (_Float16)(xj - xi);
  }
}

__global__ void gc_zero(float* __restrict__ ws) {
  if (threadIdx.x < 2 * COUT) ws[threadIdx.x] = 0.0f;
}

// Pass 1: recompute-free stats pass. Accumulate per-channel sum(h), sum(h^2).
__global__ __launch_bounds__(128) void gc_stats(const float* __restrict__ x,
                                                const float* __restrict__ w,
                                                const int* __restrict__ ei,
                                                float* __restrict__ ws) {
  __shared__ _Float16 featLDS[NT * FSTR];
  int blk  = blockIdx.x;
  int b    = blk >> 9;            // N_/NT = 512 chunks per batch
  int n0   = (blk & 511) * NT;
  int tid  = threadIdx.x;
  int wave = tid >> 5;
  int lane = tid & 31;

  v16h a0 = load_a_frag(w, wave, lane, 0);
  v16h a1 = load_a_frag(w, wave, lane, 1);
  v16h a2 = load_a_frag(w, wave, lane, 2);
  v16h a3 = load_a_frag(w, wave, lane, 3);

  float sum[8], sq[8];
#pragma unroll
  for (int v = 0; v < 8; ++v) { sum[v] = 0.0f; sq[v] = 0.0f; }

  for (int t = 0; t < TILES; ++t) {
    __syncthreads();                       // protect LDS reuse across tiles
    gather_tile(featLDS, x, ei, b, n0, t, tid);
    __syncthreads();
    v8f acc = {};
    acc = wmma_f16(a0, load_b_frag(featLDS, lane, 0), acc);
    acc = wmma_f16(a1, load_b_frag(featLDS, lane, 1), acc);
    acc = wmma_f16(a2, load_b_frag(featLDS, lane, 2), acc);
    acc = wmma_f16(a3, load_b_frag(featLDS, lane, 3), acc);
#pragma unroll
    for (int v = 0; v < 8; ++v) {
      float h = acc[v];
      sum[v] += h;
      sq[v]  += h * h;
    }
  }

  // Butterfly-reduce across the 16 lanes that share each channel (wave32).
#pragma unroll
  for (int v = 0; v < 8; ++v) {
    float s1 = sum[v], s2 = sq[v];
#pragma unroll
    for (int msk = 8; msk >= 1; msk >>= 1) {
      s1 += __shfl_xor(s1, msk, 32);
      s2 += __shfl_xor(s2, msk, 32);
    }
    if ((lane & 15) == 0) {
      int c = wave * 16 + ((lane < 16) ? v : 8 + v);
      atomicAdd(&ws[c], s1);
      atomicAdd(&ws[COUT + c], s2);
    }
  }
}

__global__ void gc_finalize(const float* __restrict__ gamma,
                            const float* __restrict__ beta,
                            float* __restrict__ ws) {
  int c = threadIdx.x;
  if (c < COUT) {
    const float S = (float)B_ * (float)N_ * (float)K_;
    float mean = ws[c] / S;
    float var  = ws[COUT + c] / S - mean * mean;
    var = fmaxf(var, 0.0f);
    float inv = rsqrtf(var + 1e-5f);
    float sc  = gamma[c] * inv;
    ws[2 * COUT + c] = sc;                  // scale
    ws[3 * COUT + c] = beta[c] - mean * sc; // bias
  }
}

// Pass 2: recompute h, apply BN scale/bias + ReLU, max over K, write output.
__global__ __launch_bounds__(128) void gc_out(const float* __restrict__ x,
                                              const float* __restrict__ w,
                                              const int* __restrict__ ei,
                                              const float* __restrict__ ws,
                                              float* __restrict__ out) {
  __shared__ _Float16 featLDS[NT * FSTR];
  __shared__ float outBuf[NT * COUT];
  int blk  = blockIdx.x;
  int b    = blk >> 9;
  int n0   = (blk & 511) * NT;
  int tid  = threadIdx.x;
  int wave = tid >> 5;
  int lane = tid & 31;

  for (int i = tid; i < NT * COUT; i += 128) outBuf[i] = 0.0f;

  v16h a0 = load_a_frag(w, wave, lane, 0);
  v16h a1 = load_a_frag(w, wave, lane, 1);
  v16h a2 = load_a_frag(w, wave, lane, 2);
  v16h a3 = load_a_frag(w, wave, lane, 3);

  float sc[8], bi[8];
#pragma unroll
  for (int v = 0; v < 8; ++v) {
    int c = wave * 16 + ((lane < 16) ? v : 8 + v);
    sc[v] = ws[2 * COUT + c];
    bi[v] = ws[3 * COUT + c];
  }

  for (int t = 0; t < TILES; ++t) {
    __syncthreads();
    gather_tile(featLDS, x, ei, b, n0, t, tid);
    __syncthreads();
    v8f acc = {};
    acc = wmma_f16(a0, load_b_frag(featLDS, lane, 0), acc);
    acc = wmma_f16(a1, load_b_frag(featLDS, lane, 1), acc);
    acc = wmma_f16(a2, load_b_frag(featLDS, lane, 2), acc);
    acc = wmma_f16(a3, load_b_frag(featLDS, lane, 3), acc);

    int s    = t * 16 + (lane & 15);
    int nloc = s / K_;
#pragma unroll
    for (int v = 0; v < 8; ++v) {
      int c = wave * 16 + ((lane < 16) ? v : 8 + v);
      float val = fmaxf(acc[v] * sc[v] + bi[v], 0.0f);
      // post-ReLU values are >= 0, so IEEE order == int order: ds_max_i32.
      atomicMax((int*)&outBuf[nloc * COUT + c], __float_as_int(val));
    }
  }
  __syncthreads();
  // out[b][c][n0+nloc]; 16 consecutive threads share a channel -> 64B bursts.
  for (int i = tid; i < NT * COUT; i += 128) {
    int c    = i >> 4;
    int nloc = i & 15;
    out[((size_t)b * COUT + c) * N_ + n0 + nloc] = outBuf[nloc * COUT + c];
  }
}

extern "C" void kernel_launch(void* const* d_in, const int* in_sizes, int n_in,
                              void* d_out, int out_size, void* d_ws, size_t ws_size,
                              hipStream_t stream) {
  (void)in_sizes; (void)n_in; (void)out_size; (void)ws_size;
  const float* x     = (const float*)d_in[0];   // [8,64,8192,1]
  const float* w     = (const float*)d_in[1];   // [64,128]
  const float* gamma = (const float*)d_in[2];   // [64]
  const float* beta  = (const float*)d_in[3];   // [64]
  const int*   ei    = (const int*)d_in[4];     // [2,8,8192,20]
  float* ws  = (float*)d_ws;                    // [0:64)=sum [64:128)=sumsq
  float* out = (float*)d_out;                   // [8,64,8192]

  dim3 grid(B_ * (N_ / NT));                    // 4096 workgroups
  gc_zero<<<1, 128, 0, stream>>>(ws);
  gc_stats<<<grid, 128, 0, stream>>>(x, w, ei, ws);
  gc_finalize<<<1, 64, 0, stream>>>(gamma, beta, ws);
  gc_out<<<grid, 128, 0, stream>>>(x, w, ei, ws, out);
}
